// BasicFeatureMineBlock_58531814309933
// MI455X (gfx1250) — matrix operand order repaired and checked by reading, not compiled, not run on hardware
//
#include <hip/hip_runtime.h>
#include <math.h>

// ---------------- problem constants ----------------
constexpr int kB  = 4;
constexpr int kC  = 48;
constexpr int kH  = 192;
constexpr int kW  = 144;
constexpr int kHW = kH * kW;          // 27648
constexpr int kNPIX = kB * kHW;       // 110592
constexpr int kOG = 6;
constexpr int kCg = 8;                // kC / kOG
constexpr int kCOUT = 6;
constexpr int kKC = 14;               // K chunks: 432 -> 448 = 14*32

typedef __attribute__((ext_vector_type(16))) _Float16 v16h;
typedef __attribute__((ext_vector_type(8)))  float    v8f;

// =====================================================================
// SE attention: global-average-pool
// =====================================================================
__global__ __launch_bounds__(256) void k_pool(const float* __restrict__ x,
                                              float* __restrict__ p) {
  const int bc = blockIdx.x;                  // 0..191  (b*48+c)
  const float* base = x + (size_t)bc * kHW;
  float s = 0.f;
  for (int i = threadIdx.x; i < kHW; i += 256) s += base[i];
  __shared__ float r[256];
  r[threadIdx.x] = s;
  __syncthreads();
  for (int st = 128; st > 0; st >>= 1) {
    if (threadIdx.x < st) r[threadIdx.x] += r[threadIdx.x + st];
    __syncthreads();
  }
  if (threadIdx.x == 0) p[bc] = r[0] * (1.f / (float)kHW);
}

// SE FC layers: s = sigmoid(relu(p@W1^T+b1)@W2^T+b2)   (tiny: one block)
__global__ __launch_bounds__(128) void k_se(const float* __restrict__ p,
                                            const float* __restrict__ w1,
                                            const float* __restrict__ b1,
                                            const float* __restrict__ w2,
                                            const float* __restrict__ b2,
                                            float* __restrict__ sbuf) {
  __shared__ float sp[kB * kC];       // 192
  __shared__ float sh[kB * 2 * kC];   // 384
  for (int i = threadIdx.x; i < kB * kC; i += 128) sp[i] = p[i];
  __syncthreads();
  for (int i = threadIdx.x; i < kB * 2 * kC; i += 128) {
    int b = i / (2 * kC), j = i % (2 * kC);
    float a = b1[j];
    for (int c = 0; c < kC; c++) a += sp[b * kC + c] * w1[j * kC + c];
    sh[i] = fmaxf(a, 0.f);
  }
  __syncthreads();
  for (int i = threadIdx.x; i < kB * kC; i += 128) {
    int b = i / kC, c = i % kC;
    float a = b2[c];
    for (int j = 0; j < 2 * kC; j++) a += sh[b * 2 * kC + j] * w2[c * 2 * kC + j];
    sbuf[i] = 1.f / (1.f + expf(-a));
  }
}

// h0 = x * s ; also emit f16 copy for the WMMA conv
__global__ __launch_bounds__(256) void k_scale(const float* __restrict__ x,
                                               const float* __restrict__ sbuf,
                                               float* __restrict__ h32,
                                               _Float16* __restrict__ h16) {
  int i = blockIdx.x * 256 + threadIdx.x;     // n = kNPIX*kC, exact multiple
  float v = x[i] * sbuf[i / kHW];
  h32[i] = v;
  h16[i] = (_Float16)v;
}

// =====================================================================
// Pack conv weights (Cout x 48 x 3 x 3 fp32) into a per-lane
// A-fragment layout for V_WMMA_F32_16X16X32_F16 (contiguous k order, same as B).
//   k = c*9 + tap, K padded to 448 (14 chunks of 32), M padded to 16*Mtiles.
//   A half i at lane L of chunk kc: k = kc*32 + i + (L>>4)*16
// =====================================================================
__global__ __launch_bounds__(128) void k_pack(const float* __restrict__ w,
                                              _Float16* __restrict__ A,
                                              int Cout, int Mtiles) {
  int idx = blockIdx.x * 128 + threadIdx.x;       // one unit = (mt,kc,lane)
  int total = Mtiles * kKC * 32;
  if (idx >= total) return;
  int lane = idx & 31;
  int kc = (idx >> 5) % kKC;
  int mt = idx / (kKC * 32);
  int m = mt * 16 + (lane & 15);
  _Float16* dst = A + (size_t)idx * 16;           // ((mt*14+kc)*32+lane)*16
  #pragma unroll
  for (int i = 0; i < 16; i++) {
    int k = kc * 32 + i + ((lane >> 4) << 4);
    int c = k / 9, t = k % 9;
    float v = (m < Cout && c < kC) ? w[((size_t)m * kC + c) * 9 + t] : 0.f;
    dst[i] = (_Float16)v;
  }
}

// =====================================================================
// Implicit-GEMM 3x3 conv via V_WMMA_F32_16X16X32_F16, Cin = 48, pad == dil.
// Workgroup: 3 waves, one (batch,row, 48-px strip); each wave owns 16 px.
// LDS holds im2col tile sB[48 px][448 k] in f16 so a B fragment is one
// contiguous 32B LDS read per lane (contiguous k order, same as A: k=(L/16)*16+i).
// mode: 0 = BN+ReLU -> f16 only
//       1 = BN+res+ReLU -> f32 + f16
//       2 = bias -> f32
//       3 = bias+sigmoid -> f32
// =====================================================================
template <int MT, int DIL>
__global__ void __launch_bounds__(96)
k_conv48_wmma(const _Float16* __restrict__ Aw, const _Float16* __restrict__ X,
              float* __restrict__ out32, _Float16* __restrict__ out16,
              const float* __restrict__ p0, const float* __restrict__ p1,
              const float* __restrict__ res, int Cout, int mode) {
  __shared__ _Float16 sB[48 * 448];   // 43008 B
  const int tid = threadIdx.x;
  const int xg = blockIdx.x, y = blockIdx.y, b = blockIdx.z;
  const int x0 = xg * 48;

  // ---- stage im2col tile: pair (x local, c) -> 9 taps ----
  for (int pr = tid; pr < 48 * kC; pr += 96) {
    int x = pr % 48, c = pr / 48;
    const _Float16* src = X + ((size_t)(b * kC + c)) * kHW;
    _Float16* d = &sB[x * 448 + c * 9];
    #pragma unroll
    for (int t = 0; t < 9; t++) {
      int yy = y + (t / 3 - 1) * DIL;
      int xx = x0 + x + (t % 3 - 1) * DIL;
      _Float16 v = (_Float16)0.f;
      if (yy >= 0 && yy < kH && xx >= 0 && xx < kW) v = src[yy * kW + xx];
      d[t] = v;
    }
  }
  if (tid < 48) {           // zero-pad k = 432..447
    #pragma unroll
    for (int kk = 432; kk < 448; kk++) sB[tid * 448 + kk] = (_Float16)0.f;
  }
  __syncthreads();

  const int lane = tid & 31, wv = tid >> 5;
  v8f acc[MT];
  #pragma unroll
  for (int m = 0; m < MT; m++) acc[m] = (v8f){0.f,0.f,0.f,0.f,0.f,0.f,0.f,0.f};

  const int colLocal = wv * 16 + (lane & 15);
  const int krow = (lane >> 4) * 16;

  for (int kc = 0; kc < kKC; kc++) {
    // B fragment: 16 contiguous halves (32B, 2x ds_load_b128)
    v16h bb = *(const v16h*)&sB[colLocal * 448 + kc * 32 + krow];
    if (kc + 1 < kKC)  // warm L0 for next A chunk (global_prefetch_b8)
      __builtin_prefetch((const void*)(Aw + ((size_t)((kc + 1) * 32 + lane)) * 16), 0, 1);
    #pragma unroll
    for (int m = 0; m < MT; m++) {
      v16h a = *(const v16h*)(Aw + ((size_t)((m * kKC + kc) * 32 + lane)) * 16);
      acc[m] = __builtin_amdgcn_wmma_f32_16x16x32_f16(
          false, a, false, bb, (short)0, acc[m], false, false);
    }
  }

  // ---- fused epilogue + store (C/D layout: cout = mt*16+r+(lane>=16)*8) ----
  const int xcol = x0 + colLocal;
  #pragma unroll
  for (int m = 0; m < MT; m++) {
    #pragma unroll
    for (int r = 0; r < 8; r++) {
      int cout = m * 16 + r + (lane >> 4) * 8;
      if (cout >= Cout) continue;
      size_t idx = (((size_t)(b * Cout + cout)) * kH + y) * kW + xcol;
      float av = acc[m][r];
      if (mode == 0) {                       // BN + ReLU -> f16
        float v = fmaxf(p0[cout] * av + p1[cout], 0.f);
        out16[idx] = (_Float16)v;
      } else if (mode == 1) {                // BN + residual + ReLU -> f32+f16
        float v = fmaxf(p0[cout] * av + p1[cout] + res[idx], 0.f);
        out32[idx] = v;
        out16[idx] = (_Float16)v;
      } else if (mode == 2) {                // bias -> f32
        out32[idx] = av + p0[cout];
      } else {                               // bias + sigmoid -> f32
        float v = av + p0[cout];
        out32[idx] = 1.f / (1.f + expf(-v));
      }
    }
  }
}

// =====================================================================
// Deformable conv (OG=6 groups, Cg=8, COUT=6): gather-bound, scalar fp32.
// msk already has sigmoid applied (fused in mask conv).
// =====================================================================
__global__ __launch_bounds__(256) void k_deform(const float* __restrict__ h,
                                                const float* __restrict__ off,
                                                const float* __restrict__ msk,
                                                const float* __restrict__ w,
                                                const float* __restrict__ bias,
                                                float* __restrict__ out) {
  __shared__ float sw[kCOUT * kC * 9];      // 2592
  for (int i = threadIdx.x; i < kCOUT * kC * 9; i += 256) sw[i] = w[i];
  __syncthreads();
  int pix = blockIdx.x * 256 + threadIdx.x; // kNPIX exact multiple
  int b = pix / kHW, rem = pix % kHW, y = rem / kW, x = rem % kW;
  float acc[kCOUT] = {0.f, 0.f, 0.f, 0.f, 0.f, 0.f};
  for (int g = 0; g < kOG; g++) {
    for (int k = 0; k < 9; k++) {
      int ky = k / 3, kx = k % 3;
      size_t ob = (((size_t)(b * 108 + (g * 9 + k) * 2)) * kH + y) * kW + x;
      float dy = off[ob];
      float dx = off[ob + (size_t)kHW];
      float m = msk[(((size_t)(b * 54 + g * 9 + k)) * kH + y) * kW + x];
      float py = (float)(y - 3 + ky * 3) + dy;
      float px = (float)(x - 3 + kx * 3) + dx;
      float y0f = floorf(py), x0f = floorf(px);
      float wy1 = py - y0f, wx1 = px - x0f;
      int y0 = (int)y0f, xx0 = (int)x0f;
      float w00 = (1.f - wy1) * (1.f - wx1), w01 = (1.f - wy1) * wx1;
      float w10 = wy1 * (1.f - wx1), w11 = wy1 * wx1;
      bool v0y = (y0 >= 0) && (y0 < kH), v1y = (y0 + 1 >= 0) && (y0 + 1 < kH);
      bool v0x = (xx0 >= 0) && (xx0 < kW), v1x = (xx0 + 1 >= 0) && (xx0 + 1 < kW);
      #pragma unroll
      for (int c = 0; c < kCg; c++) {
        const float* cb = h + ((size_t)(b * kC + g * kCg + c)) * kHW;
        float v00 = (v0y && v0x) ? cb[y0 * kW + xx0] : 0.f;
        float v01 = (v0y && v1x) ? cb[y0 * kW + xx0 + 1] : 0.f;
        float v10 = (v1y && v0x) ? cb[(y0 + 1) * kW + xx0] : 0.f;
        float v11 = (v1y && v1x) ? cb[(y0 + 1) * kW + xx0 + 1] : 0.f;
        float val = (v00 * w00 + v01 * w01 + v10 * w10 + v11 * w11) * m;
        #pragma unroll
        for (int o = 0; o < kCOUT; o++)
          acc[o] += val * sw[((o * kC) + g * kCg + c) * 9 + k];
      }
    }
  }
  #pragma unroll
  for (int o = 0; o < kCOUT; o++)
    out[(((size_t)(b * kCOUT + o)) * kH + y) * kW + x] = acc[o] + bias[o];
}

// =====================================================================
// Direct 3x3 conv for 6 channels + fused BN (+residual) + ReLU, fp32.
// =====================================================================
__global__ __launch_bounds__(256) void k_conv6(const float* __restrict__ in,
                                               const float* __restrict__ w,
                                               const float* __restrict__ gamma,
                                               const float* __restrict__ beta,
                                               const float* __restrict__ res,
                                               float* __restrict__ out) {
  __shared__ float sw[kCOUT * kCOUT * 9];   // 324
  for (int i = threadIdx.x; i < kCOUT * kCOUT * 9; i += 256) sw[i] = w[i];
  __syncthreads();
  int pix = blockIdx.x * 256 + threadIdx.x;
  int b = pix / kHW, rem = pix % kHW, y = rem / kW, x = rem % kW;
  float acc[kCOUT] = {0.f, 0.f, 0.f, 0.f, 0.f, 0.f};
  #pragma unroll
  for (int t = 0; t < 9; t++) {
    int yy = y + t / 3 - 1, xx = x + t % 3 - 1;
    if (yy < 0 || yy >= kH || xx < 0 || xx >= kW) continue;
    size_t base = (((size_t)b * kCOUT) * kH + yy) * kW + xx;
    #pragma unroll
    for (int ci = 0; ci < kCOUT; ci++) {
      float v = in[base + (size_t)ci * kHW];
      #pragma unroll
      for (int co = 0; co < kCOUT; co++) acc[co] += v * sw[(co * kCOUT + ci) * 9 + t];
    }
  }
  #pragma unroll
  for (int co = 0; co < kCOUT; co++) {
    float r = gamma[co] * acc[co] + beta[co];
    size_t idx = (((size_t)(b * kCOUT + co)) * kH + y) * kW + x;
    if (res) r += res[idx];
    out[idx] = fmaxf(r, 0.f);
  }
}

// =====================================================================
// Host launcher
// =====================================================================
extern "C" void kernel_launch(void* const* d_in, const int* in_sizes, int n_in,
                              void* d_out, int out_size, void* d_ws, size_t ws_size,
                              hipStream_t stream) {
  const float* x      = (const float*)d_in[0];
  const float* se_w1  = (const float*)d_in[1];
  const float* se_b1  = (const float*)d_in[2];
  const float* se_w2  = (const float*)d_in[3];
  const float* se_b2  = (const float*)d_in[4];
  const float* sp_w1  = (const float*)d_in[5];
  const float* sp_g1  = (const float*)d_in[6];
  const float* sp_be1 = (const float*)d_in[7];
  const float* sp_w2  = (const float*)d_in[8];
  const float* sp_g2  = (const float*)d_in[9];
  const float* sp_be2 = (const float*)d_in[10];
  const float* off_w  = (const float*)d_in[11];
  const float* off_b  = (const float*)d_in[12];
  const float* msk_w  = (const float*)d_in[13];
  const float* msk_b  = (const float*)d_in[14];
  const float* dcn_w  = (const float*)d_in[15];
  const float* dcn_b  = (const float*)d_in[16];
  const float* pp1_w1 = (const float*)d_in[17];
  const float* pp1_g1 = (const float*)d_in[18];
  const float* pp1_be1= (const float*)d_in[19];
  const float* pp1_w2 = (const float*)d_in[20];
  const float* pp1_g2 = (const float*)d_in[21];
  const float* pp1_be2= (const float*)d_in[22];
  const float* pp2_w1 = (const float*)d_in[23];
  const float* pp2_g1 = (const float*)d_in[24];
  const float* pp2_be1= (const float*)d_in[25];
  const float* pp2_w2 = (const float*)d_in[26];
  const float* pp2_g2 = (const float*)d_in[27];
  const float* pp2_be2= (const float*)d_in[28];

  // ---- workspace carve-up (256B aligned) ----
  char* ws = (char*)d_ws;
  size_t cur = 0;
  auto alloc = [&](size_t bytes) -> char* {
    char* p = ws + cur;
    cur = (cur + bytes + 255) & ~(size_t)255;
    return p;
  };
  const size_t nAct = (size_t)kNPIX * kC;       // 5,308,416
  float*    p_pool = (float*)alloc(192 * 4);
  float*    sbuf   = (float*)alloc(192 * 4);
  float*    h0_32  = (float*)alloc(nAct * 4);
  _Float16* h0_16  = (_Float16*)alloc(nAct * 2);
  _Float16* h1_16  = (_Float16*)alloc(nAct * 2);
  float*    h2_32  = (float*)alloc(nAct * 4);
  _Float16* h2_16  = (_Float16*)alloc(nAct * 2);
  float*    off32  = (float*)alloc((size_t)kNPIX * 108 * 4);
  float*    msk32  = (float*)alloc((size_t)kNPIX * 54 * 4);
  _Float16* Aw1    = (_Float16*)alloc((size_t)3 * kKC * 512 * 2);
  _Float16* Aw2    = (_Float16*)alloc((size_t)3 * kKC * 512 * 2);
  _Float16* AwO    = (_Float16*)alloc((size_t)7 * kKC * 512 * 2);
  _Float16* AwM    = (_Float16*)alloc((size_t)4 * kKC * 512 * 2);
  float*    dcn    = (float*)alloc((size_t)kNPIX * kCOUT * 4);
  float*    t1     = (float*)alloc((size_t)kNPIX * kCOUT * 4);
  float*    t2     = (float*)alloc((size_t)kNPIX * kCOUT * 4);

  // ---- SE attention ----
  k_pool<<<kB * kC, 256, 0, stream>>>(x, p_pool);
  k_se<<<1, 128, 0, stream>>>(p_pool, se_w1, se_b1, se_w2, se_b2, sbuf);
  k_scale<<<(int)(nAct / 256), 256, 0, stream>>>(x, sbuf, h0_32, h0_16);

  // ---- pack weights into WMMA A-fragment layout ----
  k_pack<<<(3 * kKC * 32 + 127) / 128, 128, 0, stream>>>(sp_w1, Aw1, 48, 3);
  k_pack<<<(3 * kKC * 32 + 127) / 128, 128, 0, stream>>>(sp_w2, Aw2, 48, 3);
  k_pack<<<(7 * kKC * 32 + 127) / 128, 128, 0, stream>>>(off_w, AwO, 108, 7);
  k_pack<<<(4 * kKC * 32 + 127) / 128, 128, 0, stream>>>(msk_w, AwM, 54, 4);

  dim3 cg(kW / 48, kH, kB);   // 3 x 192 x 4 workgroups, 96 threads each
  // spatial basic block (48ch) via WMMA implicit GEMM
  k_conv48_wmma<3, 1><<<cg, 96, 0, stream>>>(Aw1, h0_16, nullptr, h1_16,
                                             sp_g1, sp_be1, nullptr, 48, 0);
  k_conv48_wmma<3, 1><<<cg, 96, 0, stream>>>(Aw2, h1_16, h2_32, h2_16,
                                             sp_g2, sp_be2, h0_32, 48, 1);
  // dilated offset / mask convs (dil=pad=3)
  k_conv48_wmma<7, 3><<<cg, 96, 0, stream>>>(AwO, h2_16, off32, nullptr,
                                             off_b, nullptr, nullptr, 108, 2);
  k_conv48_wmma<4, 3><<<cg, 96, 0, stream>>>(AwM, h2_16, msk32, nullptr,
                                             msk_b, nullptr, nullptr, 54, 3);

  // ---- deformable conv (gather-bound, fp32) ----
  k_deform<<<kNPIX / 256, 256, 0, stream>>>(h2_32, off32, msk32, dcn_w, dcn_b, dcn);

  // ---- two 6-channel basic blocks ----
  k_conv6<<<kNPIX / 256, 256, 0, stream>>>(dcn, pp1_w1, pp1_g1, pp1_be1, nullptr, t1);
  k_conv6<<<kNPIX / 256, 256, 0, stream>>>(t1, pp1_w2, pp1_g2, pp1_be2, dcn, t2);
  k_conv6<<<kNPIX / 256, 256, 0, stream>>>(t2, pp2_w1, pp2_g1, pp2_be1, nullptr, t1);
  k_conv6<<<kNPIX / 256, 256, 0, stream>>>(t1, pp2_w2, pp2_g2, pp2_be2, t2,
                                           (float*)d_out);
  (void)in_sizes; (void)n_in; (void)out_size; (void)ws_size;
}
